// VolumeProjector_14448269984207
// MI455X (gfx1250) — matrix operand order repaired and checked by reading, not compiled, not run on hardware
//
#include <hip/hip_runtime.h>

// ---------------------------------------------------------------------------
// Problem constants (from the reference)
// ---------------------------------------------------------------------------
#define B_       2
#define NCAM     6
#define NIMG     12            // B_*NCAM
#define CBEV     192
#define DBINS    50
#define IH       16
#define IW       44
#define NP       704           // IH*IW
#define ENC      9600          // CBEV*DBINS
#define OC1      256
#define OC2      88
#define VX       100           // voxel X (vol W)
#define VY       100           // voxel Y (vol H)
#define VZ       8             // voxel Z (vol D)

#define IC_BLK   320           // K elements staged per LDS round (10 WMMA steps)
#define SUBK     (IC_BLK / 32)

typedef __bf16 bf16;
typedef __attribute__((ext_vector_type(8)))  __bf16 bf16x8;
typedef __attribute__((ext_vector_type(16))) __bf16 bf16x16;
typedef __attribute__((ext_vector_type(8)))  float  f32x8;
typedef __attribute__((ext_vector_type(4)))  unsigned int u32x4;
typedef __attribute__((ext_vector_type(8)))  int          i32x8;
typedef __attribute__((ext_vector_type(4)))  int          i32x4;

#if defined(__has_builtin)
#if __has_builtin(__builtin_amdgcn_tensor_load_to_lds)
#define USE_TDM 1
#endif
#endif
#ifndef USE_TDM
#define USE_TDM 0
#endif

// ---------------------------------------------------------------------------
// Workspace layout (bytes)
// ---------------------------------------------------------------------------
static constexpr size_t OFF_MATS = 0;                                   // 12*32 floats
static constexpr size_t OFF_W1   = 1536;
static constexpr size_t SZ_W1    = (size_t)9 * OC1 * ENC * 2;           // 44,236,800
static constexpr size_t OFF_W2   = OFF_W1 + SZ_W1;
static constexpr size_t SZ_W2    = (size_t)9 * OC2 * OC1 * 2;           // 405,504
static constexpr size_t OFF_X1   = OFF_W2 + SZ_W2;
static constexpr size_t SZ_X1    = (size_t)NIMG * NP * OC1 * 2;         // 4,325,376
static constexpr size_t OFF_S    = OFF_X1 + SZ_X1;

// ---------------------------------------------------------------------------
// 3x3 inverse (adjugate)
// ---------------------------------------------------------------------------
__device__ inline void inv3(const float* m, float* o) {
  float a=m[0],b=m[1],c=m[2],d=m[3],e=m[4],f=m[5],g=m[6],h=m[7],i=m[8];
  float A = e*i - f*h;
  float Bc = -(d*i - f*g);
  float Cc = d*h - e*g;
  float det = a*A + b*Bc + c*Cc;
  float r = 1.0f / det;
  o[0] =  A * r;            o[1] = -(b*i - c*h) * r;  o[2] =  (b*f - c*e) * r;
  o[3] =  Bc * r;           o[4] =  (a*i - c*g) * r;  o[5] = -(a*f - c*d) * r;
  o[6] =  Cc * r;           o[7] = -(a*h - b*g) * r;  o[8] =  (a*e - b*d) * r;
}

// ---------------------------------------------------------------------------
// Kernel 1: per-(b,n) matrices
// ---------------------------------------------------------------------------
__global__ void geom_kernel(const float* __restrict__ s2e,
                            const float* __restrict__ intrins,
                            const float* __restrict__ post_rots,
                            const float* __restrict__ bda,
                            float* __restrict__ mats) {
  int t = threadIdx.x;
  if (t >= NIMG) return;
  int b = t / NCAM;
  float* rec = mats + t * 32;
  inv3(post_rots + t * 9, rec + 0);
  inv3(intrins   + t * 9, rec + 9);
  const float* S = s2e + t * 16;
  const float* Bm = bda + b * 9;
  for (int i = 0; i < 3; ++i)
    for (int j = 0; j < 4; ++j) {
      float acc = 0.f;
      for (int k = 0; k < 3; ++k) acc += Bm[i*3+k] * S[k*4+j];
      rec[18 + i*4 + j] = acc;
    }
}

// ---------------------------------------------------------------------------
// Kernel 2: project + trilinear sample -> s_bf16 [img][h][w][ic], ic = c*50+d
// ---------------------------------------------------------------------------
__global__ void sample_kernel(const float* __restrict__ bev,
                              const float* __restrict__ post_trans,
                              const float* __restrict__ mats,
                              unsigned short* __restrict__ s_out_u16) {
  bf16* s_out = reinterpret_cast<bf16*>(s_out_u16);
  int idx = blockIdx.x * blockDim.x + threadIdx.x;
  if (idx >= NIMG * DBINS * NP) return;
  int w = idx % IW;   int t = idx / IW;
  int h = t % IH;     t /= IH;
  int d = t % DBINS;  t /= DBINS;
  int n = t % NCAM;
  int b = t / NCAM;
  int img = b * NCAM + n;

  int f  = h * IW + w;
  int xi = f >> 4;
  int yi = f & 15;
  float px = (float)xi * (703.0f / 43.0f);
  float py = (float)yi * 17.0f;
  float dep = 1.0f + (float)d * (44.0f / 49.0f);

  const float* pt  = post_trans + img * 3;
  const float* rec = mats + img * 32;

  float vx = px - pt[0], vy = py - pt[1], vz = dep - pt[2];
  const float* iPR = rec + 0;
  float u0 = iPR[0]*vx + iPR[1]*vy + iPR[2]*vz;
  float u1 = iPR[3]*vx + iPR[4]*vy + iPR[5]*vz;
  float u2 = iPR[6]*vx + iPR[7]*vy + iPR[8]*vz;
  float c0 = u0 * u2, c1 = u1 * u2, c2 = u2;
  const float* iK = rec + 9;
  float q0 = iK[0]*c0 + iK[1]*c1 + iK[2]*c2;
  float q1 = iK[3]*c0 + iK[4]*c1 + iK[5]*c2;
  float q2 = iK[6]*c0 + iK[7]*c1 + iK[8]*c2;
  const float* M = rec + 18;
  float ex = M[0]*q0 + M[1]*q1 + M[2]*q2  + M[3];
  float ey = M[4]*q0 + M[5]*q1 + M[6]*q2  + M[7];
  float ez = M[8]*q0 + M[9]*q1 + M[10]*q2 + M[11];

  float xv = (ex + 40.0f) * (99.0f / 80.0f);
  float yv = (ey + 40.0f) * (99.0f / 80.0f);
  float zv = (ez + 1.0f)  * (7.0f  / 6.4f);

  float x0f = floorf(xv), y0f = floorf(yv), z0f = floorf(zv);
  float fx = xv - x0f, fy = yv - y0f, fz = zv - z0f;
  int x0 = (int)x0f, y0 = (int)y0f, z0 = (int)z0f;

  size_t off[8];
  float  wt[8];
  int k = 0;
  for (int cz = 0; cz < 2; ++cz)
    for (int cy = 0; cy < 2; ++cy)
      for (int cx = 0; cx < 2; ++cx, ++k) {
        int ix = x0 + cx, iy = y0 + cy, iz = z0 + cz;
        float wgt = (cx ? fx : 1.f - fx) * (cy ? fy : 1.f - fy) * (cz ? fz : 1.f - fz);
        bool valid = (unsigned)ix < VX && (unsigned)iy < VY && (unsigned)iz < VZ;
        int cxi = ix < 0 ? 0 : (ix > VX-1 ? VX-1 : ix);
        int cyi = iy < 0 ? 0 : (iy > VY-1 ? VY-1 : iy);
        int czi = iz < 0 ? 0 : (iz > VZ-1 ? VZ-1 : iz);
        off[k] = (size_t)cxi * 800 + (size_t)cyi * 8 + (size_t)czi;
        wt[k]  = valid ? wgt : 0.0f;
      }

  const float* volb = bev + (size_t)b * CBEV * 80000;
  size_t sbase = ((size_t)(img * IH + h) * IW + w) * ENC + d;
  for (int c = 0; c < CBEV; ++c) {
    const float* vp = volb + (size_t)c * 80000;
    float acc = 0.f;
#pragma unroll
    for (int j = 0; j < 8; ++j) acc += wt[j] * vp[off[j]];
    s_out[sbase + (size_t)c * DBINS] = (bf16)acc;
  }
}

// ---------------------------------------------------------------------------
// Kernel 3/4: cast conv weights f32 [OC][IC][3][3] -> bf16 [tap][oc][ic]
// ---------------------------------------------------------------------------
__global__ void castw_kernel(const float* __restrict__ w,
                             unsigned short* __restrict__ wt_u16,
                             int OC, int IC) {
  bf16* wt = reinterpret_cast<bf16*>(wt_u16);
  long long g = (long long)blockIdx.x * blockDim.x + threadIdx.x;
  long long tot = (long long)OC * IC * 9;
  if (g >= tot) return;
  int kx = (int)(g % 3); long long t = g / 3;
  int ky = (int)(t % 3); t /= 3;
  int ic = (int)(t % IC);
  int oc = (int)(t / IC);
  int tap = ky * 3 + kx;
  wt[((size_t)tap * OC + oc) * IC + ic] = (bf16)w[g];
}

// ---------------------------------------------------------------------------
// WMMA fragment helpers (wave32, 16x16x32 bf16)
// ---------------------------------------------------------------------------
__device__ inline bf16x16 pack16(bf16x8 lo, bf16x8 hi) {
  return __builtin_shufflevector(lo, hi, 0,1,2,3,4,5,6,7,8,9,10,11,12,13,14,15);
}

// ---------------------------------------------------------------------------
// TDM: DMA one 2D tile (tile1 rows x tile0 bf16 elements, row stride in
// elements) from global into LDS. Descriptor per CDNA5 ISA §8 (D# groups).
// 6-arg toolchain form: (g0 v4u32, g1 v8i32, g2 v4i32, g3 v4i32, gx v8i32, cpol)
// ---------------------------------------------------------------------------
__device__ inline void tdm_load_2d(unsigned int lds_addr, const void* gptr,
                                   unsigned int tile0, unsigned int tile1,
                                   unsigned long long stride0_elems) {
#if USE_TDM
  unsigned long long ga = (unsigned long long)(uintptr_t)gptr;
  const unsigned int TD0 = 1u << 30;   // tensor_dim0 (OOB bound; tiles stay in range)
  const unsigned int TD1 = 1u << 20;   // tensor_dim1
  u32x4 g0;
  g0[0] = 1u;                                               // count=1, user mode
  g0[1] = lds_addr;                                         // LDS byte address
  g0[2] = (unsigned int)ga;                                 // global addr [31:0]
  g0[3] = ((unsigned int)(ga >> 32) & 0x01FFFFFFu)
        | (2u << 30);                                       // addr[56:32] | type=2
  i32x8 g1;
  g1[0] = (int)(1u << 16);                                  // data_size=1 (2 bytes)
  g1[1] = (int)((TD0 & 0xFFFFu) << 16);                     // tensor_dim0 lo16
  g1[2] = (int)((TD0 >> 16) | ((TD1 & 0xFFFFu) << 16));     // td0 hi | td1 lo
  g1[3] = (int)((TD1 >> 16) | (tile0 << 16));               // td1 hi | tile_dim0
  g1[4] = (int)(tile1 & 0xFFFFu);                           // tile_dim1 (tile_dim2=0)
  g1[5] = (int)(unsigned int)(stride0_elems & 0xFFFFFFFFu); // dim0 stride lo32
  g1[6] = (int)((unsigned int)(stride0_elems >> 32) & 0xFFFFu); // stride hi16
  g1[7] = 0;
  i32x4 g2; g2[0] = 1; g2[1] = 1; g2[2] = 0; g2[3] = 0;     // dims 2/3 unused
  i32x4 g3; g3[0] = 0; g3[1] = 0; g3[2] = 0; g3[3] = 0;
  i32x8 gx;                                                 // trailing group (unused)
  gx[0] = 0; gx[1] = 0; gx[2] = 0; gx[3] = 0;
  gx[4] = 0; gx[5] = 0; gx[6] = 0; gx[7] = 0;
  __builtin_amdgcn_tensor_load_to_lds(g0, g1, g2, g3, gx, 0);
#else
  (void)lds_addr; (void)gptr; (void)tile0; (void)tile1; (void)stride0_elems;
#endif
}

// ---------------------------------------------------------------------------
// Kernel 5: conv1 implicit GEMM.
// 256 thr = 8 waves = 4 Mtiles x 2 Ntiles; block tile 64 oc x 32 positions.
// K = 9 taps x 9600 ic, staged in IC_BLK=320 chunks (10 WMMA per stage).
// A tile loaded by the Tensor Data Mover (1 TDM per wave, 8 rows each).
// ---------------------------------------------------------------------------
__global__ void __launch_bounds__(256)
conv1_kernel(const unsigned short* __restrict__ w1t_u16,
             const unsigned short* __restrict__ s_u16,
             const float* __restrict__ bias1,
             unsigned short* __restrict__ x1_u16) {
  const bf16* w1t = reinterpret_cast<const bf16*>(w1t_u16);
  const bf16* s   = reinterpret_cast<const bf16*>(s_u16);
  bf16* x1        = reinterpret_cast<bf16*>(x1_u16);

  __shared__ __align__(16) bf16 Alds[64 * IC_BLK];  // 40 KB
  __shared__ __align__(16) bf16 Blds[32 * IC_BLK];  // 20 KB

  const int img = blockIdx.z;
  const int ocb = blockIdx.y * 64;
  const int nb  = blockIdx.x * 32;

  const int tid  = threadIdx.x;
  const int wid  = tid >> 5;
  const int lane = tid & 31;
  const int mi = wid >> 1;        // 0..3
  const int ni = wid & 1;         // 0..1
  const int half = lane >> 4;
  const int lo16 = lane & 15;

  // B staging role: n = tid>>3 (0..31), 4-elem (8B) columns
  const int b_n   = tid >> 3;
  const int b_ic4 = (tid & 7) * 4;
  const int p_b  = nb + b_n;
  const int hh   = p_b / IW;
  const int ww   = p_b % IW;

#if !USE_TDM
  const int a_oc  = tid >> 2;            // 0..63
  const int a_ic8 = (tid & 3) * 8;
#endif

  const unsigned int ldsA =
      (unsigned int)(uintptr_t)&Alds[0] + (unsigned int)(wid * 8 * IC_BLK * 2);

  f32x8 acc = {};

  for (int tap = 0; tap < 9; ++tap) {
    const int dy = tap / 3 - 1;
    const int dx = tap % 3 - 1;
    const int hy = hh + dy;
    const int wx = ww + dx;
    const bool vB = (unsigned)hy < IH && (unsigned)wx < IW;
    const bf16* sB = vB ? (s + ((size_t)(img * IH + hy) * IW + wx) * ENC) : s;

    for (int icb = 0; icb < ENC; icb += IC_BLK) {
      __syncthreads();   // previous-stage LDS reads complete before overwrite

#if USE_TDM
      // A: each wave DMAs its 8 rows of the 64 x IC_BLK weight tile
      const bf16* gpA = w1t + ((size_t)tap * OC1 + ocb + wid * 8) * ENC + icb;
      tdm_load_2d(ldsA, gpA, IC_BLK, 8, (unsigned long long)ENC);
#else
      const bf16* wA = w1t + ((size_t)tap * OC1 + ocb + a_oc) * ENC + icb;
#pragma unroll
      for (int j = 0; j < SUBK; ++j) {
        *reinterpret_cast<uint4*>(&Alds[a_oc * IC_BLK + a_ic8 + j * 32]) =
            *reinterpret_cast<const uint4*>(wA + a_ic8 + j * 32);
      }
#endif
      // B: per-thread VMEM loads (exact SAME-padding zero fill)
      __builtin_prefetch(reinterpret_cast<const char*>(sB + icb) + IC_BLK * 2, 0, 1);
#pragma unroll
      for (int j = 0; j < SUBK; ++j) {
        uint2 bval = make_uint2(0u, 0u);
        if (vB) bval = *reinterpret_cast<const uint2*>(sB + icb + b_ic4 + j * 32);
        *reinterpret_cast<uint2*>(&Blds[b_n * IC_BLK + b_ic4 + j * 32]) = bval;
      }
#if USE_TDM
      __builtin_amdgcn_s_wait_tensorcnt(0);
#endif
      __syncthreads();

      const bf16* ap = &Alds[(mi * 16 + lo16) * IC_BLK];
      const bf16* bp = &Blds[(ni * 16 + lo16) * IC_BLK];
#pragma unroll
      for (int k = 0; k < SUBK; ++k) {
        const int ic = k * 32;
        bf16x8 a0 = *reinterpret_cast<const bf16x8*>(ap + ic + 8 * half);
        bf16x8 a1 = *reinterpret_cast<const bf16x8*>(ap + ic + 16 + 8 * half);
        bf16x8 b0 = *reinterpret_cast<const bf16x8*>(bp + ic + 8 * half);
        bf16x8 b1 = *reinterpret_cast<const bf16x8*>(bp + ic + 16 + 8 * half);
        acc = __builtin_amdgcn_wmma_f32_16x16x32_bf16(
            false, pack16(a0, a1), false, pack16(b0, b1),
            (short)0, acc, false, false);
      }
    }
  }

  // epilogue: bias + relu + bf16, one 16B store per lane
  const int p   = nb + ni * 16 + lo16;
  const int oc0 = ocb + mi * 16 + 8 * half;
  __align__(16) bf16 ov[8];
#pragma unroll
  for (int r = 0; r < 8; ++r) {
    float v = acc[r] + bias1[oc0 + r];
    ov[r] = (bf16)fmaxf(v, 0.0f);
  }
  *reinterpret_cast<uint4*>(&x1[((size_t)img * NP + p) * OC1 + oc0]) =
      *reinterpret_cast<const uint4*>(ov);
}

// ---------------------------------------------------------------------------
// Kernel 6: conv2 implicit GEMM. 8 waves = 2 Mtiles x 4 Ntiles.
// Block tile: 32 oc x 64 positions. K = 9 taps x 256 ic. M masked at 88.
// ---------------------------------------------------------------------------
__global__ void __launch_bounds__(256)
conv2_kernel(const unsigned short* __restrict__ w2t_u16,
             const unsigned short* __restrict__ x1_u16,
             const float* __restrict__ bias2,
             float* __restrict__ out) {
  const bf16* w2t = reinterpret_cast<const bf16*>(w2t_u16);
  const bf16* x1  = reinterpret_cast<const bf16*>(x1_u16);

  __shared__ __align__(16) bf16 A2[32 * 32];
  __shared__ __align__(16) bf16 B2[64 * 32];

  const int img = blockIdx.z;
  const int ocb = blockIdx.y * 32;
  const int nb  = blockIdx.x * 64;

  const int tid  = threadIdx.x;
  const int wid  = tid >> 5;
  const int lane = tid & 31;
  const int mi = wid >> 2;
  const int ni = wid & 3;
  const int half = lane >> 4;
  const int lo16 = lane & 15;

  const int a_oc  = tid >> 2;
  const int a_ic8 = (tid & 3) * 8;
  const int b_n   = tid >> 2;
  const int b_ic8 = (tid & 3) * 8;

  const int p_b = nb + b_n;
  const int hh  = p_b / IW;
  const int ww  = p_b % IW;

  f32x8 acc = {};

  for (int tap = 0; tap < 9; ++tap) {
    const int dy = tap / 3 - 1;
    const int dx = tap % 3 - 1;
    const int hy = hh + dy;
    const int wx = ww + dx;
    const bool vB = (unsigned)hy < IH && (unsigned)wx < IW;
    const bf16* xB = vB ? (x1 + ((size_t)img * NP + hy * IW + wx) * OC1) : x1;
    const int ocA = ocb + a_oc;
    const bool vA = (tid < 128) && (ocA < OC2);
    const bf16* wA = vA ? (w2t + ((size_t)tap * OC2 + ocA) * OC1) : w2t;

    for (int ic0 = 0; ic0 < OC1; ic0 += 32) {
      __syncthreads();
      if (tid < 128) {
        uint4 aval = make_uint4(0u, 0u, 0u, 0u);
        if (vA) aval = *reinterpret_cast<const uint4*>(wA + ic0 + a_ic8);
        *reinterpret_cast<uint4*>(&A2[a_oc * 32 + a_ic8]) = aval;
      }
      uint4 bval = make_uint4(0u, 0u, 0u, 0u);
      if (vB) bval = *reinterpret_cast<const uint4*>(xB + ic0 + b_ic8);
      *reinterpret_cast<uint4*>(&B2[b_n * 32 + b_ic8]) = bval;
      __syncthreads();

      const bf16* ap = &A2[(mi * 16 + lo16) * 32];
      bf16x8 a0 = *reinterpret_cast<const bf16x8*>(ap + 8 * half);
      bf16x8 a1 = *reinterpret_cast<const bf16x8*>(ap + 16 + 8 * half);
      const bf16* bp = &B2[(ni * 16 + lo16) * 32 + 8 * half];
      bf16x8 b0 = *reinterpret_cast<const bf16x8*>(bp);
      bf16x8 b1 = *reinterpret_cast<const bf16x8*>(bp + 16);
      acc = __builtin_amdgcn_wmma_f32_16x16x32_bf16(
          false, pack16(a0, a1), false, pack16(b0, b1),
          (short)0, acc, false, false);
    }
  }

  const int p   = nb + ni * 16 + lo16;
  const int h   = p / IW;
  const int w   = p % IW;
  const int oc0 = ocb + mi * 16 + 8 * half;
#pragma unroll
  for (int r = 0; r < 8; ++r) {
    int oc = oc0 + r;
    if (oc < OC2) {
      float v = fmaxf(acc[r] + bias2[oc], 0.0f);
      out[((size_t)(img * OC2 + oc) * IH + h) * IW + w] = v;
    }
  }
}

// ---------------------------------------------------------------------------
// Host launch
// ---------------------------------------------------------------------------
extern "C" void kernel_launch(void* const* d_in, const int* in_sizes, int n_in,
                              void* d_out, int out_size, void* d_ws, size_t ws_size,
                              hipStream_t stream) {
  const float* bev  = (const float*)d_in[0];
  const float* s2e  = (const float*)d_in[2];
  const float* intr = (const float*)d_in[4];
  const float* prot = (const float*)d_in[5];
  const float* ptr_ = (const float*)d_in[6];
  const float* bda  = (const float*)d_in[7];
  const float* w1   = (const float*)d_in[8];
  const float* b1   = (const float*)d_in[9];
  const float* w2   = (const float*)d_in[10];
  const float* b2   = (const float*)d_in[11];

  char* ws = (char*)d_ws;
  float*          mats  = (float*)(ws + OFF_MATS);
  unsigned short* w1t   = (unsigned short*)(ws + OFF_W1);
  unsigned short* w2t   = (unsigned short*)(ws + OFF_W2);
  unsigned short* x1    = (unsigned short*)(ws + OFF_X1);
  unsigned short* sfeat = (unsigned short*)(ws + OFF_S);
  float* out = (float*)d_out;

  geom_kernel<<<1, 16, 0, stream>>>(s2e, intr, prot, bda, mats);

  {
    long long tot = (long long)OC1 * ENC * 9;     // 22,118,400
    castw_kernel<<<(unsigned)((tot + 255) / 256), 256, 0, stream>>>(w1, w1t, OC1, ENC);
  }
  {
    long long tot = (long long)OC2 * OC1 * 9;     // 202,752
    castw_kernel<<<(unsigned)((tot + 255) / 256), 256, 0, stream>>>(w2, w2t, OC2, OC1);
  }

  {
    int npts = NIMG * DBINS * NP;                 // 422,400
    sample_kernel<<<(npts + 255) / 256, 256, 0, stream>>>(bev, ptr_, mats, sfeat);
  }

  conv1_kernel<<<dim3(NP / 32, OC1 / 64, NIMG), 256, 0, stream>>>(w1t, sfeat, b1, x1);
  conv2_kernel<<<dim3(NP / 64, 3, NIMG), 256, 0, stream>>>(w2t, x1, b2, out);
}
